// BindNode23SAGEConvMLP_62715112456264
// MI455X (gfx1250) — compile-verified, matrix operand order checked by reading, not run on hardware
//
#include <hip/hip_runtime.h>

typedef __attribute__((ext_vector_type(16))) _Float16 v16h;
typedef __attribute__((ext_vector_type(8)))  _Float16 v8h;
typedef __attribute__((ext_vector_type(8)))  float    v8f;

#define N_NODES   70000
#define N_EDGES   800000
#define IN_SAGE   1024
#define FEAT_CH   128
#define FEAT_RAW  1044
#define EXTRA     20
#define CONC      (FEAT_CH + EXTRA)   /* 148 */
#define HIDDEN    37
#define OUTC      3
#define KTOT      2048                /* combined [mean | x_in] K dim */
#define XSTRIDE   152                 /* padded row stride for x tile */
#define HSTRIDE   40

// ---------------- zero scratch (agg + cnt) ----------------
__global__ void k_zero(float* __restrict__ agg, unsigned* __restrict__ cnt) {
  long i = (long)blockIdx.x * blockDim.x + threadIdx.x;
  long stride = (long)gridDim.x * blockDim.x;
  const long n_agg = (long)N_NODES * IN_SAGE;
  for (long j = i; j < n_agg; j += stride) agg[j] = 0.0f;
  for (long j = i; j < N_NODES; j += stride) cnt[j] = 0u;
}

// ---------------- convert & fuse weights: wc[n][k], k<1024 -> Wl, else Wr ---
__global__ void k_wconv(const float* __restrict__ wl, const float* __restrict__ wr,
                        _Float16* __restrict__ wc) {
  int i = blockIdx.x * blockDim.x + threadIdx.x;
  if (i >= FEAT_CH * KTOT) return;
  int n = i / KTOT, k = i % KTOT;
  float v = (k < IN_SAGE) ? wl[n * IN_SAGE + k] : wr[n * IN_SAGE + (k - IN_SAGE)];
  wc[i] = (_Float16)v;
}

// ---------------- degree count ----------------
__global__ void k_count(const int* __restrict__ edges, unsigned* __restrict__ cnt) {
  int e = blockIdx.x * blockDim.x + threadIdx.x;
  if (e < N_EDGES) atomicAdd(&cnt[edges[N_EDGES + e]], 1u);
}

// ---------------- edge scatter: agg[dst] += x_in[src] (one block per edge) --
__global__ __launch_bounds__(256) void k_scatter(const float* __restrict__ feat,
                                                 const int* __restrict__ edges,
                                                 float* __restrict__ agg) {
  int e = blockIdx.x;
  int src = edges[e];
  int dst = edges[N_EDGES + e];
  const float4* fr = (const float4*)(feat + (size_t)src * FEAT_RAW);
  float4 v = fr[threadIdx.x];                       // 256 lanes * float4 = 1024 floats
  float* o = agg + (size_t)dst * IN_SAGE + threadIdx.x * 4;
  atomicAdd(o + 0, v.x);
  atomicAdd(o + 1, v.y);
  atomicAdd(o + 2, v.z);
  atomicAdd(o + 3, v.w);
}

// ---------------- mean = agg / max(cnt,1) (in place) ----------------
__global__ void k_norm(float* __restrict__ agg, const unsigned* __restrict__ cnt) {
  long i = (long)blockIdx.x * blockDim.x + threadIdx.x;
  if (i >= (long)N_NODES * IN_SAGE) return;
  unsigned c = cnt[i >> 10];                        // IN_SAGE == 1024
  agg[i] *= 1.0f / (float)((c == 0u) ? 1u : c);
}

// ---------------- fused WMMA GEMM + concat + fc1 + BN + fc2 ----------------
__global__ __launch_bounds__(256) void k_fused(
    const float* __restrict__ feat, const float* __restrict__ meanbuf,
    const _Float16* __restrict__ wc, const float* __restrict__ b_sage,
    const float* __restrict__ fc1_w, const float* __restrict__ fc1_b,
    const float* __restrict__ fc2_w, const float* __restrict__ fc2_b,
    const float* __restrict__ bn_g, const float* __restrict__ bn_b,
    const float* __restrict__ bn_m, const float* __restrict__ bn_v,
    float* __restrict__ out) {
  extern __shared__ char smem[];
  _Float16* a_lds = (_Float16*)smem;                                  // 16 x 2048 f16 (64KB)
  float*    x_lds = (float*)(smem + (size_t)16 * KTOT * 2);           // 16 x 152 f32
  float*    h_lds = x_lds + 16 * XSTRIDE;                             // 16 x 40  f32

  const int tid   = threadIdx.x;
  const int node0 = blockIdx.x * 16;

  // Stage combined A tile [mean | x_in] for 16 nodes into LDS as f16.
  for (int i = tid; i < 16 * KTOT; i += 256) {
    int r = i >> 11;                 // / 2048
    int c = i & (KTOT - 1);
    int node = node0 + r;
    float v = (c < IN_SAGE) ? meanbuf[(size_t)node * IN_SAGE + c]
                            : feat[(size_t)node * FEAT_RAW + (c - IN_SAGE)];
    a_lds[i] = (_Float16)v;
  }
  __syncthreads();

  const int lane = tid & 31;
  const int wv   = tid >> 5;        // wave 0..7 -> output-channel tile
  const int m16  = lane & 15;
  const int kh   = lane >> 4;       // 0/1
  const int n0   = wv * 16;
  const int ncol = n0 + m16;

  v8f acc = {};
  const _Float16* arow = a_lds + m16 * KTOT;
  const _Float16* brow = wc + (size_t)ncol * KTOT;

#pragma unroll 4
  for (int kc = 0; kc < KTOT; kc += 32) {
    v8h a0 = *(const v8h*)(arow + kc + kh * 8);
    v8h a1 = *(const v8h*)(arow + kc + 16 + kh * 8);
    v8h b0 = *(const v8h*)(brow + kc + kh * 8);
    v8h b1 = *(const v8h*)(brow + kc + 16 + kh * 8);
    v16h a, b;
#pragma unroll
    for (int j = 0; j < 8; j++) {
      a[j] = a0[j]; a[j + 8] = a1[j];
      b[j] = b0[j]; b[j + 8] = b1[j];
    }
    acc = __builtin_amdgcn_wmma_f32_16x16x32_f16(
        /*neg_a=*/false, a, /*neg_b=*/false, b,
        /*c_mod=*/(short)0, acc, /*reuse_a=*/false, /*reuse_b=*/false);
  }

  // D(m,n): reg r -> m = 8*kh + r, n = ncol. Apply bias + ReLU, drop into x tile.
  float bias = b_sage[ncol];
#pragma unroll
  for (int r = 0; r < 8; r++) {
    int m = kh * 8 + r;
    float v = acc[r] + bias;
    x_lds[m * XSTRIDE + ncol] = v > 0.0f ? v : 0.0f;
  }

  // Concat the 20 extra raw feature columns.
  for (int i = tid; i < 16 * EXTRA; i += 256) {
    int r = i / EXTRA, e = i % EXTRA;
    x_lds[r * XSTRIDE + FEAT_CH + e] =
        feat[(size_t)(node0 + r) * FEAT_RAW + IN_SAGE + e];
  }
  __syncthreads();

  // fc1 (148 -> 37) + ReLU + BN.
  for (int i = tid; i < 16 * HIDDEN; i += 256) {
    int r = i / HIDDEN, h = i % HIDDEN;
    float s = fc1_b[h];
    const float* wrow = fc1_w + h * CONC;
    for (int j = 0; j < CONC; j++) s += wrow[j] * x_lds[r * XSTRIDE + j];
    s = s > 0.0f ? s : 0.0f;
    s = (s - bn_m[h]) * rsqrtf(bn_v[h] + 1e-5f) * bn_g[h] + bn_b[h];
    h_lds[r * HSTRIDE + h] = s;
  }
  __syncthreads();

  // fc2 (37 -> 3), write output.
  for (int i = tid; i < 16 * OUTC; i += 256) {
    int r = i / OUTC, o = i % OUTC;
    float s = fc2_b[o];
    const float* wrow = fc2_w + o * HIDDEN;
    for (int j = 0; j < HIDDEN; j++) s += wrow[j] * h_lds[r * HSTRIDE + j];
    out[(size_t)(node0 + r) * OUTC + o] = s;
  }
}

extern "C" void kernel_launch(void* const* d_in, const int* in_sizes, int n_in,
                              void* d_out, int out_size, void* d_ws, size_t ws_size,
                              hipStream_t stream) {
  (void)in_sizes; (void)n_in; (void)out_size; (void)ws_size;
  const float* feat  = (const float*)d_in[0];
  const int*   edges = (const int*)d_in[1];
  // d_in[2] (edges2) and d_in[3] (edge_features) are unused by the reference.
  const float* wl    = (const float*)d_in[4];
  const float* bl    = (const float*)d_in[5];
  const float* wr    = (const float*)d_in[6];
  const float* fc1w  = (const float*)d_in[7];
  const float* fc1b  = (const float*)d_in[8];
  const float* fc2w  = (const float*)d_in[9];
  const float* fc2b  = (const float*)d_in[10];
  const float* bng   = (const float*)d_in[11];
  const float* bnb   = (const float*)d_in[12];
  const float* bnm   = (const float*)d_in[13];
  const float* bnv   = (const float*)d_in[14];
  float* out = (float*)d_out;

  char* ws = (char*)d_ws;
  float*    agg = (float*)ws;                                        // 70000*1024 f32
  unsigned* cnt = (unsigned*)(ws + (size_t)N_NODES * IN_SAGE * 4);   // 70000 u32
  _Float16* wcb = (_Float16*)(ws + (size_t)N_NODES * IN_SAGE * 4 + (size_t)N_NODES * 4);

  k_zero<<<4096, 256, 0, stream>>>(agg, cnt);
  k_wconv<<<(FEAT_CH * KTOT + 255) / 256, 256, 0, stream>>>(wl, wr, wcb);
  k_count<<<(N_EDGES + 255) / 256, 256, 0, stream>>>(edges, cnt);
  k_scatter<<<N_EDGES, 256, 0, stream>>>(feat, edges, agg);
  k_norm<<<((long)N_NODES * IN_SAGE + 255) / 256, 256, 0, stream>>>(agg, cnt);

  size_t shmem = (size_t)16 * KTOT * 2 + (size_t)16 * XSTRIDE * 4 + (size_t)16 * HSTRIDE * 4;
  k_fused<<<N_NODES / 16, 256, shmem, stream>>>(feat, agg, wcb, bl,
                                                fc1w, fc1b, fc2w, fc2b,
                                                bng, bnb, bnm, bnv, out);
}
